// Generator_10101763080572
// MI455X (gfx1250) — matrix operand order repaired and checked by reading, not compiled, not run on hardware
//
#include <hip/hip_runtime.h>
#include <hip/hip_bf16.h>

// ---------------------------------------------------------------------------
// StyleGAN2-style generator forward for MI455X (gfx1250, wave32, WMMA).
//
// Modulated conv:  y = demod[b,o] * ( W (conv) (x * style[b,i]) )
// => stage x*style once per conv as f16 in CHANNEL-LAST layout [b][pixel][i],
// run a shared-weight implicit GEMM with v_wmma_f32_16x16x32_f16.
// Each wave register-blocks 2x2 tiles (32 out-ch x 32 pixels): per K-chunk it
// loads 2 A-fragments + 2 B-fragments (all K-contiguous b128 loads) and
// issues 4 WMMAs, halving fragment traffic per FLOP vs 1 tile/wave.
// Demod folds into the f32 epilogue.  The 2x up-conv path reuses the kernel
// with parity-masked zero-dilated source coords + flipped taps, then a fused
// blur+noise+bias+leakyReLU kernel.
// ---------------------------------------------------------------------------

typedef __attribute__((ext_vector_type(16))) _Float16 v16h;
typedef __attribute__((ext_vector_type(8)))  _Float16 v8h;
typedef __attribute__((ext_vector_type(8)))  float    v8f;

#define LATENT   512
#define BATCH    8
#define SQRT2    1.41421356237309515f
#define INV_SQRT_LAT 0.04419417382415922f   /* sqrt(1/512) */
#define EPS      1e-8f

__device__ __forceinline__ float lrelu_g(float v) {
    return (v > 0.f ? v : 0.2f * v) * SQRT2;
}

// ---------------------------------------------------------------------------
// Mapping network: pixel-norm + 8 equalized FC (lr=0.01) with leaky*sqrt2.
// One block per batch sample, 512 threads, activations updated in place in one LDS buffer.
// ---------------------------------------------------------------------------
struct MapPtrs { const float* w[8]; const float* b[8]; };

__global__ void k_mapping(const float* __restrict__ z, MapPtrs mp,
                          float* __restrict__ lat) {
    __shared__ float sh[LATENT];
    const int b = blockIdx.x, t = threadIdx.x;
    float v = z[b * LATENT + t];
    sh[t] = v * v;
    __syncthreads();
    for (int s = LATENT / 2; s > 0; s >>= 1) {
        if (t < s) sh[t] += sh[t + s];
        __syncthreads();
    }
    const float scale = rsqrtf(sh[0] * (1.0f / LATENT) + EPS);
    __syncthreads();
    sh[t] = v * scale;
    __syncthreads();
    for (int l = 0; l < 8; ++l) {
        const float* W  = mp.w[l];
        const float* Bb = mp.b[l];
        float acc = 0.f;
        #pragma unroll 4
        for (int i = 0; i < LATENT; ++i) acc += sh[i] * W[t * LATENT + i];
        float y = acc * (INV_SQRT_LAT * 0.01f) + Bb[t] * 0.01f;
        y = lrelu_g(y);
        __syncthreads();
        sh[t] = y;
        __syncthreads();
    }
    lat[b * LATENT + t] = sh[t];
}

// ---------------------------------------------------------------------------
// style[b,i] = (latent[b] . aff_w[i]) * sqrt(1/512) + aff_b[i], times `post`
// ---------------------------------------------------------------------------
__global__ void k_style(const float* __restrict__ lat,
                        const float* __restrict__ aw,
                        const float* __restrict__ ab,
                        float* __restrict__ st, int I, float post, int N) {
    int idx = blockIdx.x * blockDim.x + threadIdx.x;
    if (idx >= N) return;
    int b = idx / I, i = idx % I;
    float acc = 0.f;
    #pragma unroll 4
    for (int l = 0; l < LATENT; ++l) acc += lat[b * LATENT + l] * aw[i * LATENT + l];
    st[idx] = (acc * INV_SQRT_LAT + ab[i]) * post;
}

// S[o,i] = sum_taps w^2
__global__ void k_wsumsq(const float* __restrict__ w, float* __restrict__ S,
                         int I, int N) {
    int idx = blockIdx.x * blockDim.x + threadIdx.x;
    if (idx >= N) return;
    const float* p = w + (size_t)idx * 9;
    float s = 0.f;
    #pragma unroll
    for (int t = 0; t < 9; ++t) s += p[t] * p[t];
    S[idx] = s;
}

// demod[b,o] = rsqrt( sum_i S[o,i]*style[b,i]^2 + eps )
__global__ void k_demod(const float* __restrict__ S, const float* __restrict__ st,
                        float* __restrict__ dm, int I, int O, int N) {
    int idx = blockIdx.x * blockDim.x + threadIdx.x;
    if (idx >= N) return;
    int b = idx / O, o = idx % O;
    float acc = 0.f;
    #pragma unroll 4
    for (int i = 0; i < I; ++i) {
        float s = st[b * I + i];
        acc += S[o * I + i] * s * s;
    }
    dm[idx] = rsqrtf(acc + EPS);
}

// fp32 weights [O][I][3][3] -> f16 [tap][O][I] (K contiguous for WMMA A frags)
__global__ void k_w2h(const float* __restrict__ w, _Float16* __restrict__ wh,
                      int I, int O, int N) {
    int idx = blockIdx.x * blockDim.x + threadIdx.x;
    if (idx >= N) return;
    int o = idx / (I * 9), r = idx % (I * 9), i = r / 9, t = r % 9;
    wh[((size_t)t * O + o) * I + i] = (_Float16)w[idx];
}

// Channel-last staging: xh[b][p][i] = f16( x[b,i,p] * style[b,i] )
__global__ void k_stage16T(const float* __restrict__ x, const float* __restrict__ st,
                           _Float16* __restrict__ xh, int I, int HW, int bcast,
                           int N) {
    int idx = blockIdx.x * blockDim.x + threadIdx.x;
    if (idx >= N) return;
    int i = idx % I, rest = idx / I, p = rest % HW, b = rest / HW;
    float xv = bcast ? x[(size_t)i * HW + p] : x[((size_t)b * I + i) * HW + p];
    xh[idx] = (_Float16)(xv * st[b * I + i]);
}

// ---------------------------------------------------------------------------
// Implicit-GEMM modulated 3x3 conv, 2x2 register blocking per wave:
// 32 out-channels x 32 pixels, 4 f32 accumulators, 4 WMMAs per K-chunk.
// 3-D grid (pixel-block, och-block, batch), 4 waves/block.
//   up==0 : normal conv, pad 1                 (OH=H)
//   up==1 : conv_transpose stride2 (pre-blur)  (OH=2H+1), flipped taps,
//           parity-masked zero-dilated source.
// ---------------------------------------------------------------------------
__global__ void __launch_bounds__(128)
k_conv_wmma(const _Float16* __restrict__ xh,      // [b][pixel][i] f16
            const _Float16* __restrict__ whT,     // [tap][o][i]   f16
            const float* __restrict__ dm,
            float* __restrict__ y,
            int I, int O, int H, int W, int OH, int OW, int up) {
    const int P = OH * OW;
    const int nBlocks = (P + 31) >> 5;            // 32 pixels per wave
    const int nb = blockIdx.x * blockDim.y + threadIdx.y;
    if (nb >= nBlocks) return;                    // wave-uniform exit
    const int ob = blockIdx.y;                    // 32 out-channels per wave
    const int b  = blockIdx.z;

    const int lane = threadIdx.x;                 // 0..31
    const int half = lane >> 4;
    const int l16  = lane & 15;
    const int o0   = ob << 5;
    const size_t HW = (size_t)H * W;

    const int n0 = (nb << 5) + l16;               // pixel columns of the 2 N-tiles
    const int n1 = n0 + 16;
    const bool np0 = n0 < P, np1 = n1 < P;
    const int oh0 = np0 ? n0 / OW : 0, ow0 = np0 ? n0 % OW : 0;
    const int oh1 = np1 ? n1 / OW : 0, ow1 = np1 ? n1 % OW : 0;

    const _Float16* xb = xh + (size_t)b * HW * I; // batch base
    const v8h zz = {};
    v8f acc00 = {}, acc01 = {}, acc10 = {}, acc11 = {};

    for (int t = 0; t < 9; ++t) {
        const int ky = t / 3, kx = t % 3;
        const int wt = up ? (8 - t) : t;          // flipped kernel on up path
        int ih0, iw0, ih1, iw1; bool ok0 = np0, ok1 = np1;
        if (up) {
            int u = oh0 + ky - 2, v = ow0 + kx - 2;
            ok0 = ok0 && u >= 0 && v >= 0 && !(u & 1) && !(v & 1);
            ih0 = u >> 1; iw0 = v >> 1; ok0 = ok0 && ih0 < H && iw0 < W;
            u = oh1 + ky - 2; v = ow1 + kx - 2;
            ok1 = ok1 && u >= 0 && v >= 0 && !(u & 1) && !(v & 1);
            ih1 = u >> 1; iw1 = v >> 1; ok1 = ok1 && ih1 < H && iw1 < W;
        } else {
            ih0 = oh0 + ky - 1; iw0 = ow0 + kx - 1;
            ok0 = ok0 && ih0 >= 0 && iw0 >= 0 && ih0 < H && iw0 < W;
            ih1 = oh1 + ky - 1; iw1 = ow1 + kx - 1;
            ok1 = ok1 && ih1 >= 0 && iw1 >= 0 && ih1 < H && iw1 < W;
        }
        const size_t poff0 = ok0 ? (size_t)(ih0 * W + iw0) * I : 0;  // clamped-safe
        const size_t poff1 = ok1 ? (size_t)(ih1 * W + iw1) * I : 0;
        const _Float16* bp0 = xb + poff0 + (half << 4);
        const _Float16* bp1 = xb + poff1 + (half << 4);
        const _Float16* ap0 = whT + ((size_t)wt * O + o0 + l16) * I + (half ? 16 : 0);
        const _Float16* ap1 = ap0 + (size_t)16 * I;

        #pragma unroll 2
        for (int c0 = 0; c0 < I; c0 += 32) {
            __builtin_prefetch(ap0 + c0 + 128, 0, 0);   // global_prefetch_b8
            // A fragments (2 o-tiles), two b128 loads each
            v8h a0lo = *(const v8h*)(ap0 + c0);
            v8h a0hi = *(const v8h*)(ap0 + c0 + 8);
            v8h a1lo = *(const v8h*)(ap1 + c0);
            v8h a1hi = *(const v8h*)(ap1 + c0 + 8);
            // B fragments (2 pixel-tiles), two b128 loads each, zero-masked
            v8h b0lo = *(const v8h*)(bp0 + c0);
            v8h b0hi = *(const v8h*)(bp0 + c0 + 8);
            v8h b1lo = *(const v8h*)(bp1 + c0);
            v8h b1hi = *(const v8h*)(bp1 + c0 + 8);
            b0lo = ok0 ? b0lo : zz;  b0hi = ok0 ? b0hi : zz;
            b1lo = ok1 ? b1lo : zz;  b1hi = ok1 ? b1hi : zz;
            v16h a0 = __builtin_shufflevector(a0lo, a0hi,
                         0,1,2,3,4,5,6,7,8,9,10,11,12,13,14,15);
            v16h a1 = __builtin_shufflevector(a1lo, a1hi,
                         0,1,2,3,4,5,6,7,8,9,10,11,12,13,14,15);
            v16h b0 = __builtin_shufflevector(b0lo, b0hi,
                         0,1,2,3,4,5,6,7,8,9,10,11,12,13,14,15);
            v16h b1 = __builtin_shufflevector(b1lo, b1hi,
                         0,1,2,3,4,5,6,7,8,9,10,11,12,13,14,15);
            acc00 = __builtin_amdgcn_wmma_f32_16x16x32_f16(
                        false, a0, false, b0, (short)0, acc00, false, false);
            acc01 = __builtin_amdgcn_wmma_f32_16x16x32_f16(
                        false, a0, false, b1, (short)0, acc01, false, false);
            acc10 = __builtin_amdgcn_wmma_f32_16x16x32_f16(
                        false, a1, false, b0, (short)0, acc10, false, false);
            acc11 = __builtin_amdgcn_wmma_f32_16x16x32_f16(
                        false, a1, false, b1, (short)0, acc11, false, false);
        }
    }
    // epilogue: demod scale + store (C layout: VGPR r -> M = r + 8*half)
    #pragma unroll
    for (int r = 0; r < 8; ++r) {
        const int olo = o0 + r + (half ? 8 : 0);
        const int ohi = olo + 16;
        const float dlo = dm[b * O + olo];
        const float dhi = dm[b * O + ohi];
        if (np0) {
            y[((size_t)b * O + olo) * P + n0] = acc00[r] * dlo;
            y[((size_t)b * O + ohi) * P + n0] = acc10[r] * dhi;
        }
        if (np1) {
            y[((size_t)b * O + olo) * P + n1] = acc01[r] * dlo;
            y[((size_t)b * O + ohi) * P + n1] = acc11[r] * dhi;
        }
    }
}

// blur(4x4, f*4) of the (2H+1)^2 up-conv output + noise + bias + leaky*sqrt2
__global__ void k_blur_act(const float* __restrict__ y, const float* __restrict__ noise,
                           const float* __restrict__ bias, const float* __restrict__ nwp,
                           float* __restrict__ out, int O, int H2, int N) {
    int idx = blockIdx.x * blockDim.x + threadIdx.x;
    if (idx >= N) return;
    const int P = H2 * H2, S1 = H2 + 1;
    int pq = idx % P, bo = idx / P, o = bo % O, b = bo / O;
    int p = pq / H2, q = pq % H2;
    const float F4[4] = {1.f, 3.f, 3.f, 1.f};
    const float* yb = y + (size_t)bo * S1 * S1;
    float acc = 0.f;
    #pragma unroll
    for (int fy = 0; fy < 4; ++fy) {
        int u = p + fy - 1;
        if (u < 0 || u > H2) continue;
        #pragma unroll
        for (int fx = 0; fx < 4; ++fx) {
            int v = q + fx - 1;
            if (v < 0 || v > H2) continue;
            acc += (F4[fy] * F4[fx] * (1.f / 16.f)) * yb[(size_t)u * S1 + v];
        }
    }
    acc += nwp[0] * noise[(size_t)b * P + pq] + bias[o];
    out[idx] = lrelu_g(acc);
}

// in-place: y = leaky(y + nw*noise + bias)*sqrt2
__global__ void k_act_noise(float* __restrict__ y, const float* __restrict__ noise,
                            const float* __restrict__ bias, const float* __restrict__ nwp,
                            int O, int P, int N) {
    int idx = blockIdx.x * blockDim.x + threadIdx.x;
    if (idx >= N) return;
    int p = idx % P, bo = idx / P, o = bo % O, b = bo / O;
    y[idx] = lrelu_g(y[idx] + nwp[0] * noise[(size_t)b * P + p] + bias[o]);
}

// toRGB: out[b,o,p] = sum_i w[o,i]*style[b,i]*x[b,i,p] + bias[o]
__global__ void k_torgb(const float* __restrict__ x, const float* __restrict__ st,
                        const float* __restrict__ w, const float* __restrict__ bias,
                        float* __restrict__ out, int I, int P, int N) {
    int idx = blockIdx.x * blockDim.x + threadIdx.x;
    if (idx >= N) return;
    int p = idx % P, bo = idx / P, o = bo % 3, b = bo / 3;
    const float* xb = x + (size_t)b * I * P + p;
    float acc = 0.f;
    #pragma unroll 4
    for (int i = 0; i < I; ++i)
        acc += w[o * I + i] * st[b * I + i] * xb[(size_t)i * P];
    out[idx] = acc + bias[o];
}

// dst += blur4x4( zero-upsample2x( src ) ), StyleGAN skip path
__global__ void k_upskip_add(const float* __restrict__ src, float* __restrict__ dst,
                             int H2, int N) {
    int idx = blockIdx.x * blockDim.x + threadIdx.x;
    if (idx >= N) return;
    const int P = H2 * H2, H = H2 >> 1;
    int pq = idx % P, bc = idx / P;
    int p = pq / H2, q = pq % H2;
    const float F4[4] = {1.f, 3.f, 3.f, 1.f};
    const float* sb = src + (size_t)bc * H * H;
    float acc = 0.f;
    #pragma unroll
    for (int fy = 0; fy < 4; ++fy) {
        int u = p + fy - 2;
        if (u < 0 || (u & 1) || (u >> 1) >= H) continue;
        #pragma unroll
        for (int fx = 0; fx < 4; ++fx) {
            int v = q + fx - 2;
            if (v < 0 || (v & 1) || (v >> 1) >= H) continue;
            acc += (F4[fy] * F4[fx] * (1.f / 16.f)) * sb[(size_t)(u >> 1) * H + (v >> 1)];
        }
    }
    dst[idx] += acc;
}

// ---------------------------------------------------------------------------
// Host-side orchestration
// ---------------------------------------------------------------------------
static inline unsigned gdim(size_t n, int blk) { return (unsigned)((n + blk - 1) / blk); }

extern "C" void kernel_launch(void* const* d_in, const int* in_sizes, int n_in,
                              void* d_out, int out_size, void* d_ws, size_t ws_size,
                              hipStream_t stream) {
    (void)in_sizes; (void)n_in; (void)out_size; (void)ws_size;
    static const int RES[7] = {4, 8, 16, 32, 64, 128, 256};
    static const int CH[7]  = {512, 512, 512, 512, 256, 128, 64};

    // ---- unpack inputs in setup_inputs() dict order -----------------------
    int idx = 0;
    const float* z = (const float*)d_in[idx++];
    MapPtrs mp;
    for (int l = 0; l < 8; ++l) {
        mp.w[l] = (const float*)d_in[idx++];
        mp.b[l] = (const float*)d_in[idx++];
    }
    const float* constp = (const float*)d_in[idx++];
    struct ConvP { const float *aw, *ab, *w, *bias, *nw; };
    struct RgbP  { const float *aw, *ab, *w, *bias; };
    ConvP c0[7], c1[7]; RgbP rg[7];
    for (int bi = 0; bi < 7; ++bi) {
        if (bi > 0) {
            c0[bi].aw = (const float*)d_in[idx++]; c0[bi].ab = (const float*)d_in[idx++];
            c0[bi].w  = (const float*)d_in[idx++]; c0[bi].bias = (const float*)d_in[idx++];
            c0[bi].nw = (const float*)d_in[idx++];
        }
        c1[bi].aw = (const float*)d_in[idx++]; c1[bi].ab = (const float*)d_in[idx++];
        c1[bi].w  = (const float*)d_in[idx++]; c1[bi].bias = (const float*)d_in[idx++];
        c1[bi].nw = (const float*)d_in[idx++];
        rg[bi].aw = (const float*)d_in[idx++]; rg[bi].ab = (const float*)d_in[idx++];
        rg[bi].w  = (const float*)d_in[idx++]; rg[bi].bias = (const float*)d_in[idx++];
    }
    const float *noi0[7] = {0}, *noi1[7] = {0};
    for (int bi = 0; bi < 7; ++bi) {
        if (bi > 0) noi0[bi] = (const float*)d_in[idx++];
        noi1[bi] = (const float*)d_in[idx++];
    }

    // ---- carve workspace ---------------------------------------------------
    char* wp = (char*)d_ws;
    auto carve = [&](size_t bytes) -> void* {
        void* r = (void*)wp; wp += (bytes + 255) & ~(size_t)255; return r;
    };
    float*    lat   = (float*)   carve((size_t)BATCH * LATENT * 4);
    float*    style = (float*)   carve((size_t)BATCH * 512 * 4);
    float*    dm    = (float*)   carve((size_t)BATCH * 512 * 4);
    float*    Ssq   = (float*)   carve((size_t)512 * 512 * 4);
    _Float16* whT   = (_Float16*)carve((size_t)512 * 512 * 9 * 2);
    _Float16* xh    = (_Float16*)carve((size_t)BATCH * 64 * 256 * 256 * 2);
    float*    actA  = (float*)   carve((size_t)BATCH * 64 * 257 * 257 * 4);
    float*    actB  = (float*)   carve((size_t)BATCH * 64 * 257 * 257 * 4);
    float*    imgA  = (float*)   carve((size_t)BATCH * 3 * 256 * 256 * 4);
    float*    imgB  = (float*)   carve((size_t)BATCH * 3 * 256 * 256 * 4);

    const int TB = 256;

    // ---- mapping network ---------------------------------------------------
    k_mapping<<<BATCH, LATENT, 0, stream>>>(z, mp, lat);

    float* cur = actA;
    float* oth = actB;
    float* imgCur = imgA;
    float* imgOth = imgB;

    for (int bi = 0; bi < 7; ++bi) {
        const int r = RES[bi], ch = CH[bi];
        const int pch = bi ? CH[bi - 1] : CH[0];

        if (bi > 0) {
            // ---------- conv0: up-conv (stride-2 transpose) + blur ---------
            const int Hs = r / 2, OHu = r + 1;
            const ConvP& c = c0[bi];
            size_t nS = (size_t)BATCH * pch;
            k_style <<<gdim(nS, TB), TB, 0, stream>>>(lat, c.aw, c.ab, style, pch, 1.f, (int)nS);
            size_t nW = (size_t)ch * pch;
            k_wsumsq<<<gdim(nW, TB), TB, 0, stream>>>(c.w, Ssq, pch, (int)nW);
            size_t nD = (size_t)BATCH * ch;
            k_demod <<<gdim(nD, TB), TB, 0, stream>>>(Ssq, style, dm, pch, ch, (int)nD);
            size_t nH = (size_t)ch * pch * 9;
            k_w2h   <<<gdim(nH, TB), TB, 0, stream>>>(c.w, whT, pch, ch, (int)nH);
            size_t nX = (size_t)BATCH * pch * Hs * Hs;
            k_stage16T<<<gdim(nX, TB), TB, 0, stream>>>(cur, style, xh, pch, Hs * Hs, 0, (int)nX);
            {
                int P = OHu * OHu, nBlocks = (P + 31) / 32;
                dim3 g(gdim(nBlocks, 4), ch / 32, BATCH), blk(32, 4);
                k_conv_wmma<<<g, blk, 0, stream>>>(xh, whT, dm, oth,
                                                   pch, ch, Hs, Hs, OHu, OHu, 1);
            }
            size_t nBl = (size_t)BATCH * ch * r * r;
            k_blur_act<<<gdim(nBl, TB), TB, 0, stream>>>(oth, noi0[bi], c.bias, c.nw,
                                                         cur, ch, r, (int)nBl);
            // x (res r, ch) now in `cur`
        }

        // ---------- conv1: normal modulated 3x3, pad 1 ---------------------
        {
            const ConvP& c = c1[bi];
            size_t nS = (size_t)BATCH * ch;
            k_style <<<gdim(nS, TB), TB, 0, stream>>>(lat, c.aw, c.ab, style, ch, 1.f, (int)nS);
            size_t nW = (size_t)ch * ch;
            k_wsumsq<<<gdim(nW, TB), TB, 0, stream>>>(c.w, Ssq, ch, (int)nW);
            size_t nD = (size_t)BATCH * ch;
            k_demod <<<gdim(nD, TB), TB, 0, stream>>>(Ssq, style, dm, ch, ch, (int)nD);
            size_t nH = (size_t)ch * ch * 9;
            k_w2h   <<<gdim(nH, TB), TB, 0, stream>>>(c.w, whT, ch, ch, (int)nH);
            size_t nX = (size_t)BATCH * ch * r * r;
            if (bi == 0)
                k_stage16T<<<gdim(nX, TB), TB, 0, stream>>>(constp, style, xh, ch, r * r, 1, (int)nX);
            else
                k_stage16T<<<gdim(nX, TB), TB, 0, stream>>>(cur, style, xh, ch, r * r, 0, (int)nX);
            {
                int P = r * r, nBlocks = (P + 31) / 32;
                dim3 g(gdim(nBlocks, 4), ch / 32, BATCH), blk(32, 4);
                k_conv_wmma<<<g, blk, 0, stream>>>(xh, whT, dm, oth,
                                                   ch, ch, r, r, r, r, 0);
            }
            size_t nA = (size_t)BATCH * ch * r * r;
            k_act_noise<<<gdim(nA, TB), TB, 0, stream>>>(oth, noi1[bi], c.bias, c.nw,
                                                         ch, r * r, (int)nA);
            float* t = cur; cur = oth; oth = t;   // x now in `cur`
        }

        // ---------- toRGB + skip pyramid -----------------------------------
        {
            const RgbP& g = rg[bi];
            size_t nS = (size_t)BATCH * ch;
            k_style<<<gdim(nS, TB), TB, 0, stream>>>(lat, g.aw, g.ab, style, ch,
                                                     sqrtf(1.f / (float)ch), (int)nS);
            float* dst = (bi == 6) ? (float*)d_out : imgOth;
            size_t nR = (size_t)BATCH * 3 * r * r;
            k_torgb<<<gdim(nR, TB), TB, 0, stream>>>(cur, style, g.w, g.bias, dst,
                                                     ch, r * r, (int)nR);
            if (bi > 0)
                k_upskip_add<<<gdim(nR, TB), TB, 0, stream>>>(imgCur, dst, r, (int)nR);
            if (bi < 6) { float* t = imgCur; imgCur = dst; imgOth = (t == dst) ? imgOth : t; }
        }
    }
}
